// BiLSTM_CRF_39178691674491
// MI455X (gfx1250) — compile-verified, matrix operand order checked by reading, not run on hardware
//
#include <hip/hip_runtime.h>
#include <hip/hip_bf16.h>

typedef __attribute__((ext_vector_type(16))) __bf16 v16bf;
typedef __attribute__((ext_vector_type(8)))  __bf16 v8bf;
typedef __attribute__((ext_vector_type(8)))  float  v8f;

#define BQ 128
#define TQ 512
#define DQ 256
#define HHQ 256
#define HDQ 512
#define NGQ 1024
#define NTQ 20
#define TAG_BEGIN 18
#define TAG_END 19

// ---------------------------------------------------------------------------
// WMMA fragment load (CDNA5 16-bit A/B layout, wave32).
// 16x32 tile from a K-major (row-major) matrix, base = &Mtx[row0][k0].
// lane<16 : row=lane,     K halves {k0+0..7}  and {k0+16..23}
// lane>=16: row=lane-16,  K halves {k0+8..15} and {k0+24..31}
// ---------------------------------------------------------------------------
__device__ __forceinline__ v16bf load_frag(const __bf16* base, int ld) {
    int lane = threadIdx.x & 31;
    int row  = lane & 15;
    int hoff = (lane >> 4) << 3;          // 0 or 8
    const __bf16* p = base + row * ld + hoff;
    v8bf lo = *(const v8bf*)(p);
    v8bf hi = *(const v8bf*)(p + 16);
    v16bf r;
#pragma unroll
    for (int i = 0; i < 8; ++i) { r[i] = lo[i]; r[8 + i] = hi[i]; }
    return r;
}

__device__ __forceinline__ float sigf(float x) { return 1.0f / (1.0f + expf(-x)); }

// ---------------------------------------------------------------------------
// Embedding gather + f32 -> bf16
// ---------------------------------------------------------------------------
__global__ void bl_embed_kernel(const int* __restrict__ sent,
                                const float* __restrict__ embed,
                                __bf16* __restrict__ x0, size_t n) {
    size_t i = (size_t)blockIdx.x * blockDim.x + threadIdx.x;
    if (i >= n) return;
    size_t bt = i >> 8;            // DQ == 256
    int d = (int)(i & 255);
    x0[i] = (__bf16)embed[(size_t)sent[bt] * DQ + d];
}

__global__ void bl_cast_kernel(const float* __restrict__ src,
                               __bf16* __restrict__ dst, size_t n) {
    size_t i = (size_t)blockIdx.x * blockDim.x + threadIdx.x;
    if (i < n) dst[i] = (__bf16)src[i];
}

__global__ void bl_addbias_kernel(const float* __restrict__ a,
                                  const float* __restrict__ b,
                                  float* __restrict__ dst, int n) {
    int i = blockIdx.x * blockDim.x + threadIdx.x;
    if (i < n) dst[i] = a[i] + b[i];
}

// dst[b][t][c] = (t < L) ? src[b][L-1-t][c] : 0   (time reversal with zero pad)
__global__ void bl_reverse_kernel(const __bf16* __restrict__ src,
                                  __bf16* __restrict__ dst,
                                  const int* __restrict__ seq_len,
                                  int C, int sstride, int dstride, size_t n) {
    size_t i = (size_t)blockIdx.x * blockDim.x + threadIdx.x;
    if (i >= n) return;
    int c = (int)(i % C);
    size_t bt = i / C;
    int t = (int)(bt % TQ);
    int b = (int)(bt / TQ);
    int L = seq_len[b];
    __bf16 v = (__bf16)0.0f;
    if (t < L) v = src[((size_t)b * TQ + (L - 1 - t)) * sstride + c];
    dst[((size_t)b * TQ + t) * dstride + c] = v;
}

// ---------------------------------------------------------------------------
// Input projection GEMM: C[M][1024] = A_bf16[M][K] * W_bf16[1024][K]^T + bias
// grid = (M/16, 1024/256), block = 128 (4 waves, each wave -> 4 N-tiles)
// ---------------------------------------------------------------------------
__global__ __launch_bounds__(128) void bl_gemm_bias_kernel(
    const __bf16* __restrict__ A, const __bf16* __restrict__ W,
    const float* __restrict__ bias, float* __restrict__ C, int K) {
    const int N = NGQ;
    int lane = threadIdx.x & 31;
    int wv   = threadIdx.x >> 5;                 // 0..3
    int m0   = blockIdx.x * 16;
    int n0   = blockIdx.y * 256 + wv * 64;

    v8f acc[4] = {};
    for (int k = 0; k < K; k += 32) {
        v16bf a = load_frag(A + (size_t)m0 * K + k, K);
        __builtin_prefetch((const void*)(W + (size_t)n0 * K + k + 32), 0, 1);
#pragma unroll
        for (int j = 0; j < 4; ++j) {
            v16bf b = load_frag(W + (size_t)(n0 + j * 16) * K + k, K);
            acc[j] = __builtin_amdgcn_wmma_f32_16x16x32_bf16(
                false, a, false, b, (short)0, acc[j], false, false);
        }
    }
    int rowoff = (lane >> 4) << 3;
    int col    = lane & 15;
#pragma unroll
    for (int j = 0; j < 4; ++j) {
        int n = n0 + j * 16 + col;
        float bn = bias[n];
#pragma unroll
        for (int r = 0; r < 8; ++r)
            C[(size_t)(m0 + r + rowoff) * N + n] = acc[j][r] + bn;
    }
}

// ---------------------------------------------------------------------------
// LSTM recurrence. One block = 32 batch rows for the whole sequence.
// 8 waves; wave w owns hidden columns [w*32, w*32+32) and computes all 4
// gates for them (N-tiles at q*256 + j0), so no cross-wave gate exchange.
// h ping-pongs in LDS (bf16); c lives in registers (stable lane mapping).
// W_hh (bf16, 512KB) streams from L2 every step.
// ---------------------------------------------------------------------------
#define LSTM_MB 32
__global__ __launch_bounds__(256) void bl_lstm_kernel(
    const float* __restrict__ xp,        // [B][T][1024], biases folded in
    const __bf16* __restrict__ whh,      // [1024][256] bf16
    const int* __restrict__ seq_len,
    __bf16* __restrict__ y,              // out base, per-(b,t) row stride = ystride
    int ystride) {
    __shared__ __bf16 h_s[2][LSTM_MB][HHQ];
    __shared__ int sl[LSTM_MB];

    int tid  = threadIdx.x;
    int lane = tid & 31;
    int wv   = tid >> 5;                  // 0..7
    int mb   = blockIdx.x * LSTM_MB;

    if (tid < LSTM_MB) sl[tid] = seq_len[mb + tid];
    for (int i = tid; i < 2 * LSTM_MB * HHQ; i += 256)
        ((__bf16*)h_s)[i] = (__bf16)0.0f;
    __syncthreads();

    float cst[2][2][8];
#pragma unroll
    for (int a = 0; a < 2; ++a)
#pragma unroll
        for (int b = 0; b < 2; ++b)
#pragma unroll
            for (int r = 0; r < 8; ++r) cst[a][b][r] = 0.0f;

    int rowoff = (lane >> 4) << 3;
    int col    = lane & 15;

    for (int t = 0; t < TQ; ++t) {
        int cur = t & 1, nxt = cur ^ 1;
#pragma unroll 1
        for (int mt = 0; mt < 2; ++mt) {
            v8f acc[2][4] = {};
            for (int k = 0; k < HHQ; k += 32) {
                v16bf a = load_frag(&h_s[cur][mt * 16][0] + k, HHQ);
#pragma unroll
                for (int jt = 0; jt < 2; ++jt) {
                    int j0 = wv * 32 + jt * 16;
#pragma unroll
                    for (int q = 0; q < 4; ++q) {
                        v16bf b = load_frag(whh + (size_t)(q * 256 + j0) * HHQ + k, HHQ);
                        acc[jt][q] = __builtin_amdgcn_wmma_f32_16x16x32_bf16(
                            false, a, false, b, (short)0, acc[jt][q], false, false);
                    }
                }
            }
            // gate nonlinearities + state update for this wave's hidden slice
#pragma unroll
            for (int jt = 0; jt < 2; ++jt) {
                int j = wv * 32 + jt * 16 + col;
#pragma unroll
                for (int r = 0; r < 8; ++r) {
                    int rl   = mt * 16 + r + rowoff;
                    int brow = mb + rl;
                    size_t xb = ((size_t)brow * TQ + t) * NGQ;
                    float gi = acc[jt][0][r] + xp[xb + j];
                    float gf = acc[jt][1][r] + xp[xb + 256 + j];
                    float gg = acc[jt][2][r] + xp[xb + 512 + j];
                    float go = acc[jt][3][r] + xp[xb + 768 + j];
                    float cold = cst[mt][jt][r];
                    float cnew = sigf(gf) * cold + sigf(gi) * tanhf(gg);
                    float hnew = sigf(go) * tanhf(cnew);
                    bool valid = t < sl[rl];
                    if (valid) cst[mt][jt][r] = cnew;
                    float hold = (float)h_s[cur][rl][j];
                    h_s[nxt][rl][j] = (__bf16)(valid ? hnew : hold);
                    y[((size_t)brow * TQ + t) * ystride + j] =
                        valid ? (__bf16)hnew : (__bf16)0.0f;
                }
            }
        }
        __syncthreads();
    }
}

// ---------------------------------------------------------------------------
// LayerNorm(512) + 20-tag projection. One wave per (b,t) row.
// ---------------------------------------------------------------------------
__global__ __launch_bounds__(256) void bl_ln_feats_kernel(
    const __bf16* __restrict__ h2, const float* __restrict__ ln_g,
    const float* __restrict__ ln_b, const float* __restrict__ out_w,
    const float* __restrict__ out_b, float* __restrict__ feats, size_t Mrows) {
    int lane = threadIdx.x & 31;
    int wv   = threadIdx.x >> 5;
    size_t row = (size_t)blockIdx.x * 8 + wv;
    if (row >= Mrows) return;
    const __bf16* hr = h2 + row * HDQ;

    float v[16];
    float s = 0.0f;
#pragma unroll
    for (int i = 0; i < 16; ++i) { v[i] = (float)hr[lane + 32 * i]; s += v[i]; }
#pragma unroll
    for (int m = 16; m; m >>= 1) s += __shfl_xor(s, m, 32);
    float mu = s * (1.0f / (float)HDQ);

    float vs = 0.0f;
#pragma unroll
    for (int i = 0; i < 16; ++i) { float d = v[i] - mu; vs += d * d; }
#pragma unroll
    for (int m = 16; m; m >>= 1) vs += __shfl_xor(vs, m, 32);
    float rstd = rsqrtf(vs * (1.0f / (float)HDQ) + 1e-5f);

    float hn[16];
#pragma unroll
    for (int i = 0; i < 16; ++i) {
        int c = lane + 32 * i;
        hn[i] = (v[i] - mu) * rstd * ln_g[c] + ln_b[c];
    }
    for (int n = 0; n < NTQ; ++n) {
        float p = 0.0f;
#pragma unroll
        for (int i = 0; i < 16; ++i) p += hn[i] * out_w[n * HDQ + lane + 32 * i];
#pragma unroll
        for (int m = 16; m; m >>= 1) p += __shfl_xor(p, m, 32);
        if (lane == 0) feats[row * NTQ + n] = p + out_b[n];
    }
}

// ---------------------------------------------------------------------------
// CRF forward (log-partition). One wave per batch row; lane n = tag.
// ---------------------------------------------------------------------------
__global__ __launch_bounds__(32) void bl_crf_kernel(
    const float* __restrict__ feats, const int* __restrict__ seq_len,
    const float* __restrict__ trans, float* __restrict__ fscore) {
    __shared__ float tr[NTQ * NTQ];
    int lane = threadIdx.x;
    int b = blockIdx.x;
    for (int i = lane; i < NTQ * NTQ; i += 32) tr[i] = trans[i];
    __syncthreads();

    int L = seq_len[b];
    int n = lane;
    float alpha = (n == TAG_BEGIN) ? 0.0f : -99999.0f;

    for (int t = 0; t < L; ++t) {
        float emit = (n < NTQ) ? feats[((size_t)b * TQ + t) * NTQ + n] : 0.0f;
        float vv[NTQ];
        float mx = -3.4e38f;
#pragma unroll
        for (int k = 0; k < NTQ; ++k) {
            float ak  = __shfl(alpha, k, 32);
            float val = ak + ((n < NTQ) ? tr[n * NTQ + k] : 0.0f);
            vv[k] = val;
            mx = fmaxf(mx, val);
        }
        float ssum = 0.0f;
#pragma unroll
        for (int k = 0; k < NTQ; ++k) ssum += expf(vv[k] - mx);
        float na = emit + mx + logf(ssum);
        if (n < NTQ) alpha = na;
    }
    float sc = alpha + ((n < NTQ) ? tr[TAG_END * NTQ + n] : 0.0f);
    float m2 = -3.4e38f;
#pragma unroll
    for (int k = 0; k < NTQ; ++k) m2 = fmaxf(m2, __shfl(sc, k, 32));
    float s2 = 0.0f;
#pragma unroll
    for (int k = 0; k < NTQ; ++k) s2 += expf(__shfl(sc, k, 32) - m2);
    if (lane == 0) fscore[b] = m2 + logf(s2);
}

__global__ void bl_gold_kernel(const int* __restrict__ tags,
                               const int* __restrict__ seq_len,
                               const float* __restrict__ feats,
                               const float* __restrict__ trans,
                               float* __restrict__ gt) {
    int b = blockIdx.x * blockDim.x + threadIdx.x;
    if (b >= BQ) return;
    int L = seq_len[b];
    int prev = TAG_BEGIN;
    float s = 0.0f;
    for (int t = 0; t < L; ++t) {
        int tg = tags[b * TQ + t];
        s += trans[tg * NTQ + prev] + feats[((size_t)b * TQ + t) * NTQ + tg];
        prev = tg;
    }
    s += trans[TAG_END * NTQ + prev];
    gt[b] = s;
}

__global__ void bl_loss_kernel(const float* __restrict__ fs,
                               const float* __restrict__ gt,
                               float* __restrict__ out) {
    __shared__ float sm[BQ];
    int i = threadIdx.x;
    sm[i] = fs[i] - gt[i];
    __syncthreads();
    for (int st = BQ / 2; st; st >>= 1) {
        if (i < st) sm[i] += sm[i + st];
        __syncthreads();
    }
    if (i == 0) out[0] = sm[0] / (float)BQ;
}

// ---------------------------------------------------------------------------
// Host launcher
// ---------------------------------------------------------------------------
extern "C" void kernel_launch(void* const* d_in, const int* in_sizes, int n_in,
                              void* d_out, int out_size, void* d_ws, size_t ws_size,
                              hipStream_t stream) {
    (void)in_sizes; (void)n_in; (void)out_size; (void)ws_size;
    const size_t M = (size_t)BQ * TQ;   // 65536 rows

    const int*   sent   = (const int*)d_in[0];
    const int*   seqlen = (const int*)d_in[1];
    const int*   tags   = (const int*)d_in[2];
    const float* embed  = (const float*)d_in[3];
    // dir-layer order: 0 = l0 fwd, 1 = l0 bwd, 2 = l1 fwd, 3 = l1 bwd
    const float* w_ih[4] = {(const float*)d_in[4],  (const float*)d_in[8],
                            (const float*)d_in[12], (const float*)d_in[16]};
    const float* w_hh[4] = {(const float*)d_in[5],  (const float*)d_in[9],
                            (const float*)d_in[13], (const float*)d_in[17]};
    const float* b_ih[4] = {(const float*)d_in[6],  (const float*)d_in[10],
                            (const float*)d_in[14], (const float*)d_in[18]};
    const float* b_hh[4] = {(const float*)d_in[7],  (const float*)d_in[11],
                            (const float*)d_in[15], (const float*)d_in[19]};
    const float* ln_g  = (const float*)d_in[20];
    const float* ln_b  = (const float*)d_in[21];
    const float* out_w = (const float*)d_in[22];
    const float* out_b = (const float*)d_in[23];
    const float* trans = (const float*)d_in[24];
    const int Kdim[4] = {DQ, DQ, HDQ, HDQ};

    char* ws = (char*)d_ws;
    size_t off = 0;
    auto alloc = [&](size_t bytes) -> void* {
        size_t o = off;
        off += (bytes + 255) & ~(size_t)255;
        return (void*)(ws + o);
    };

    __bf16* x0    = (__bf16*)alloc(M * DQ * 2);
    __bf16* xrev  = (__bf16*)alloc(M * HDQ * 2);
    float*  xp    = (float*)alloc(M * NGQ * 4);
    __bf16* hcat1 = (__bf16*)alloc(M * HDQ * 2);
    __bf16* hcat2 = (__bf16*)alloc(M * HDQ * 2);
    __bf16* hbwd  = (__bf16*)alloc(M * HHQ * 2);
    __bf16* wihb[4]; __bf16* whhb[4]; float* bias[4];
    for (int i = 0; i < 4; ++i) {
        wihb[i] = (__bf16*)alloc((size_t)NGQ * Kdim[i] * 2);
        whhb[i] = (__bf16*)alloc((size_t)NGQ * HHQ * 2);
        bias[i] = (float*)alloc((size_t)NGQ * 4);
    }
    float* feats  = (float*)alloc(M * NTQ * 4);
    float* fscore = (float*)alloc(BQ * 4);
    float* gt     = (float*)alloc(BQ * 4);

    // --- prep: embedding gather + weight casts + bias sums ---
    {
        size_t n = M * DQ;
        bl_embed_kernel<<<dim3((unsigned)((n + 255) / 256)), dim3(256), 0, stream>>>(
            sent, embed, x0, n);
    }
    for (int i = 0; i < 4; ++i) {
        size_t nw = (size_t)NGQ * Kdim[i];
        bl_cast_kernel<<<dim3((unsigned)((nw + 255) / 256)), dim3(256), 0, stream>>>(
            w_ih[i], wihb[i], nw);
        size_t nh = (size_t)NGQ * HHQ;
        bl_cast_kernel<<<dim3((unsigned)((nh + 255) / 256)), dim3(256), 0, stream>>>(
            w_hh[i], whhb[i], nh);
        bl_addbias_kernel<<<dim3(NGQ / 256), dim3(256), 0, stream>>>(
            b_ih[i], b_hh[i], bias[i], NGQ);
    }

    dim3 ggrid((unsigned)(M / 16), NGQ / 256);
    dim3 gblk(128);
    dim3 lgrid(BQ / LSTM_MB);
    dim3 lblk(256);

    // --- layer 0 ---
    bl_gemm_bias_kernel<<<ggrid, gblk, 0, stream>>>(x0, wihb[0], bias[0], xp, DQ);
    bl_lstm_kernel<<<lgrid, lblk, 0, stream>>>(xp, whhb[0], seqlen, hcat1, HDQ);
    {
        size_t n = M * DQ;
        bl_reverse_kernel<<<dim3((unsigned)((n + 255) / 256)), dim3(256), 0, stream>>>(
            x0, xrev, seqlen, DQ, DQ, DQ, n);
    }
    bl_gemm_bias_kernel<<<ggrid, gblk, 0, stream>>>(xrev, wihb[1], bias[1], xp, DQ);
    bl_lstm_kernel<<<lgrid, lblk, 0, stream>>>(xp, whhb[1], seqlen, hbwd, HHQ);
    {
        size_t n = M * HHQ;
        bl_reverse_kernel<<<dim3((unsigned)((n + 255) / 256)), dim3(256), 0, stream>>>(
            hbwd, hcat1 + HHQ, seqlen, HHQ, HHQ, HDQ, n);
    }

    // --- layer 1 ---
    bl_gemm_bias_kernel<<<ggrid, gblk, 0, stream>>>(hcat1, wihb[2], bias[2], xp, HDQ);
    bl_lstm_kernel<<<lgrid, lblk, 0, stream>>>(xp, whhb[2], seqlen, hcat2, HDQ);
    {
        size_t n = M * HDQ;
        bl_reverse_kernel<<<dim3((unsigned)((n + 255) / 256)), dim3(256), 0, stream>>>(
            hcat1, xrev, seqlen, HDQ, HDQ, HDQ, n);
    }
    bl_gemm_bias_kernel<<<ggrid, gblk, 0, stream>>>(xrev, wihb[3], bias[3], xp, HDQ);
    bl_lstm_kernel<<<lgrid, lblk, 0, stream>>>(xp, whhb[3], seqlen, hbwd, HHQ);
    {
        size_t n = M * HHQ;
        bl_reverse_kernel<<<dim3((unsigned)((n + 255) / 256)), dim3(256), 0, stream>>>(
            hbwd, hcat2 + HHQ, seqlen, HHQ, HHQ, HDQ, n);
    }

    // --- head + CRF ---
    bl_ln_feats_kernel<<<dim3((unsigned)(M / 8)), dim3(256), 0, stream>>>(
        hcat2, ln_g, ln_b, out_w, out_b, feats, M);
    bl_crf_kernel<<<dim3(BQ), dim3(32), 0, stream>>>(feats, seqlen, trans, fscore);
    bl_gold_kernel<<<dim3(1), dim3(128), 0, stream>>>(tags, seqlen, feats, trans, gt);
    bl_loss_kernel<<<dim3(1), dim3(BQ), 0, stream>>>(fscore, gt, (float*)d_out);
}
